// NodeEmbLayer_29119878266935
// MI455X (gfx1250) — matrix operand order repaired and checked by reading, not compiled, not run on hardware
//
#include <hip/hip_runtime.h>

typedef __attribute__((ext_vector_type(16))) _Float16 v16h;
typedef __attribute__((ext_vector_type(8)))  _Float16 v8h;
typedef __attribute__((ext_vector_type(8)))  float    v8f;

#define D_IN   27
#define H      256
#define K2     512
#define EPS    1e-5f
#define SLOPE  0.01f
#define LDA    40    // sA row stride in halves (bank-conflict-free, 16B multiple)
#define LDN    520   // node_in row stride in halves (260 dwords -> lane stride 4 banks)

// SLOPE in (0,1)  =>  leaky(x) == max(x, SLOPE*x)   (v_mul + v_max, no cmp/cndmask)
__device__ __forceinline__ float leaky(float x) { return fmaxf(x, SLOPE * x); }

// ---------------------------------------------------------------------------
// Pre-pack weights to f16 in the per-lane WMMA-B order this file uses (same k order as A):
//   B 32x16 (KxN) per (ntile t, kstep s): lane L -> column n=L&15,
//   K = 16*(L>=16) + i  for i=0..15 (contiguous 32B per lane).
// ---------------------------------------------------------------------------
__global__ __launch_bounds__(256) void pack_weights_kernel(
    const float* __restrict__ res_w,   // (256, 27)
    const float* __restrict__ node_w,  // (256, 512)
    _Float16* __restrict__ nodew_p,    // 16 t * 16 s * 32 lanes * 16 halves
    _Float16* __restrict__ resw_p)     // 16 t * 32 lanes * 16 halves (K padded 27->32)
{
    int tid = blockIdx.x * 256 + threadIdx.x;
    if (tid < 16 * 16 * 32) {
        int L = tid & 31, s = (tid >> 5) & 15, t = tid >> 9;
        int n = L & 15, hb = L >> 4;
        int h = t * 16 + n;
        const float* src = node_w + (size_t)h * K2 + s * 32 + hb * 16;
        v16h v;
#pragma unroll
        for (int i = 0; i < 16; ++i) v[i] = (_Float16)src[i];
        *(v16h*)(nodew_p + (size_t)tid * 16) = v;
    } else if (tid < 16 * 16 * 32 + 16 * 32) {
        int rid = tid - 16 * 16 * 32;
        int L = rid & 31, t = rid >> 5;
        int n = L & 15, hb = L >> 4;
        int h = t * 16 + n;
        v16h v;
#pragma unroll
        for (int i = 0; i < 16; ++i) {
            int d = hb * 16 + i;
            v[i] = (d < D_IN) ? (_Float16)res_w[h * D_IN + d] : (_Float16)0.0f;
        }
        *(v16h*)(resw_p + (size_t)rid * 16) = v;
    }
}

// ---------------------------------------------------------------------------
// Fused: row-norm -> GEMM1 -> LeakyReLU -> concat -> GEMM2 -> row-norm -> LeakyReLU
// One block = 16 rows, 256 threads = 8 wave32s; wave w owns N-tiles 2w, 2w+1.
// ---------------------------------------------------------------------------
__global__ __launch_bounds__(256) void node_emb_fused_kernel(
    const float* __restrict__ res_feat,   // (N, 27)
    const float* __restrict__ atom_emb,   // (N, 256)
    const _Float16* __restrict__ nodew_p,
    const _Float16* __restrict__ resw_p,
    float* __restrict__ out)              // (N, 256)
{
    __shared__ __attribute__((aligned(16))) _Float16 sA[16 * LDA];   // normalized res_feat, K padded to 32
    __shared__ __attribute__((aligned(16))) _Float16 sNI[16 * LDN];  // node_in f16: [res_emb | atom_emb]
    __shared__ float sSum[8][16];
    __shared__ float sSq[8][16];
    __shared__ float sMu[16];
    __shared__ float sRs[16];

    const int tid  = threadIdx.x;
    const int lane = tid & 31;
    const int w    = tid >> 5;
    const int n    = lane & 15;   // WMMA column / A row index for this lane
    const int hb   = lane >> 4;   // lane-half selector
    const int row0 = blockIdx.x * 16;

    // ---- Stage 1: InstanceNorm of res_feat rows (biased var), f16 into sA ----
    if (tid < 16) {
        const float* rf = res_feat + (size_t)(row0 + tid) * D_IN;
        float v[D_IN];
        float s = 0.0f, ss = 0.0f;
#pragma unroll
        for (int d = 0; d < D_IN; ++d) { float x = rf[d]; v[d] = x; s += x; ss += x * x; }
        float mu  = s * (1.0f / D_IN);
        float var = ss * (1.0f / D_IN) - mu * mu;
        float rs  = rsqrtf(var + EPS);
        _Float16* a = sA + tid * LDA;
#pragma unroll
        for (int d = 0; d < D_IN; ++d) a[d] = (_Float16)((v[d] - mu) * rs);
#pragma unroll
        for (int d = D_IN; d < 32; ++d) a[d] = (_Float16)0.0f;
    }

    // ---- Stage 2: atom_emb f32 -> f16 into right half of node_in ----
    // float4 global loads (b128), packed v8h LDS stores (2 x ds_store_b128/thread)
    {
        int m = tid >> 4, c = tid & 15;
        const float4* ae4 = (const float4*)(atom_emb + (size_t)(row0 + m) * H + c * 16);
        _Float16* dst = sNI + m * LDN + H + c * 16;
#pragma unroll
        for (int j = 0; j < 2; ++j) {
            float4 x0 = ae4[2 * j];
            float4 x1 = ae4[2 * j + 1];
            v8h hh;
            hh[0] = (_Float16)x0.x; hh[1] = (_Float16)x0.y;
            hh[2] = (_Float16)x0.z; hh[3] = (_Float16)x0.w;
            hh[4] = (_Float16)x1.x; hh[5] = (_Float16)x1.y;
            hh[6] = (_Float16)x1.z; hh[7] = (_Float16)x1.w;
            *(v8h*)(dst + 8 * j) = hh;
        }
    }
    __syncthreads();

    // ---- Stage 3: GEMM1 (16x27 x 27x256), one WMMA per N-tile ----
    v16h afrag1;
    {
        const v8h lo = *(const v8h*)&sA[n * LDA + 16 * hb];
        const v8h hi = *(const v8h*)&sA[n * LDA + 16 * hb + 8];
        afrag1 = __builtin_shufflevector(lo, hi, 0, 1, 2, 3, 4, 5, 6, 7,
                                                 8, 9, 10, 11, 12, 13, 14, 15);
    }
#pragma unroll
    for (int q = 0; q < 2; ++q) {
        int t = 2 * w + q;
        v16h b = *(const v16h*)(resw_p + (size_t)(t * 32 + lane) * 16);
        v8f c = {};
        c = __builtin_amdgcn_wmma_f32_16x16x32_f16(false, afrag1, false, b,
                                                   (short)0, c, false, false);
        // LeakyReLU, f16 into node_in columns [t*16, t*16+16)
#pragma unroll
        for (int r = 0; r < 8; ++r) {
            int m = r + 8 * hb;
            sNI[m * LDN + t * 16 + n] = (_Float16)leaky(c[r]);
        }
    }
    __syncthreads();

    // ---- Stage 4: GEMM2 (16x512 x 512x256), K loop of 16 WMMA steps ----
    const int t0 = 2 * w, t1 = 2 * w + 1;
    v8f acc0 = {}, acc1 = {};
#pragma unroll 4
    for (int s = 0; s < 16; ++s) {
        const v8h lo = *(const v8h*)&sNI[n * LDN + s * 32 + 16 * hb];
        const v8h hi = *(const v8h*)&sNI[n * LDN + s * 32 + 16 * hb + 8];
        v16h a = __builtin_shufflevector(lo, hi, 0, 1, 2, 3, 4, 5, 6, 7,
                                                 8, 9, 10, 11, 12, 13, 14, 15);
        v16h b0 = *(const v16h*)(nodew_p + (size_t)((t0 * 16 + s) * 32 + lane) * 16);
        v16h b1 = *(const v16h*)(nodew_p + (size_t)((t1 * 16 + s) * 32 + lane) * 16);
        acc0 = __builtin_amdgcn_wmma_f32_16x16x32_f16(false, a, false, b0,
                                                      (short)0, acc0, false, false);
        acc1 = __builtin_amdgcn_wmma_f32_16x16x32_f16(false, a, false, b1,
                                                      (short)0, acc1, false, false);
    }

    // ---- Stage 5: deterministic row-wise sum / sumsq reduction ----
    float sx[8], sq[8];
#pragma unroll
    for (int r = 0; r < 8; ++r) {
        float a0 = acc0[r], a1 = acc1[r];
        sx[r] = a0 + a1;
        sq[r] = a0 * a0 + a1 * a1;
    }
#pragma unroll
    for (int mask = 1; mask < 16; mask <<= 1) {
#pragma unroll
        for (int r = 0; r < 8; ++r) {
            sx[r] += __shfl_xor(sx[r], mask, 32);
            sq[r] += __shfl_xor(sq[r], mask, 32);
        }
    }
    if ((lane & 15) == 0) {
#pragma unroll
        for (int r = 0; r < 8; ++r) {
            int m = r + 8 * hb;
            sSum[w][m] = sx[r];
            sSq[w][m]  = sq[r];
        }
    }
    __syncthreads();
    if (tid < 16) {
        float s = 0.0f, ss = 0.0f;
#pragma unroll
        for (int wv = 0; wv < 8; ++wv) { s += sSum[wv][tid]; ss += sSq[wv][tid]; }
        float mu  = s * (1.0f / H);
        float var = ss * (1.0f / H) - mu * mu;
        sMu[tid] = mu;
        sRs[tid] = rsqrtf(var + EPS);
    }
    __syncthreads();

    // ---- Stage 6: normalize + LeakyReLU + store (64B-coalesced per r) ----
#pragma unroll
    for (int r = 0; r < 8; ++r) {
        int m = r + 8 * hb;
        float mu = sMu[m], rs = sRs[m];
        float y0 = leaky((acc0[r] - mu) * rs);
        float y1 = leaky((acc1[r] - mu) * rs);
        float* orow = out + (size_t)(row0 + m) * H;
        orow[t0 * 16 + n] = y0;
        orow[t1 * 16 + n] = y1;
    }
}

extern "C" void kernel_launch(void* const* d_in, const int* in_sizes, int n_in,
                              void* d_out, int out_size, void* d_ws, size_t ws_size,
                              hipStream_t stream) {
    const float* res_feat = (const float*)d_in[0];
    const float* atom_emb = (const float*)d_in[1];
    const float* res_w    = (const float*)d_in[2];
    const float* node_w   = (const float*)d_in[3];
    float* out = (float*)d_out;

    const int N = in_sizes[0] / D_IN;

    _Float16* nodew_p = (_Float16*)d_ws;                        // 256 KB
    _Float16* resw_p  = (_Float16*)((char*)d_ws + (size_t)16 * 16 * 32 * 16 * sizeof(_Float16));

    // 8192 + 512 pack threads -> 34 blocks of 256
    pack_weights_kernel<<<34, 256, 0, stream>>>(res_w, node_w, nodew_p, resw_p);
    node_emb_fused_kernel<<<N / 16, 256, 0, stream>>>(res_feat, atom_emb,
                                                      nodew_p, resw_p, out);
}
